// SimpleDeformableAggregation_32238024524442
// MI455X (gfx1250) — matrix operand order repaired and checked by reading, not compiled, not run on hardware
//
#include <hip/hip_runtime.h>
#include <hip/hip_bf16.h>
#include <stdint.h>

// ---------------- problem constants ----------------
#define B_    2
#define N_    6400
#define E_    256
#define CAMS_ 6
#define LVLS_ 4
#define ROWS  (B_ * N_)            // 12800

__constant__ int LVL_W_[4]  = {176, 88, 44, 22};
__constant__ int LVL_H_[4]  = {64, 32, 16, 8};
__constant__ int LVL_HW_[4] = {64 * 176, 32 * 88, 16 * 44, 8 * 22};

// ---------------- workspace layout (bytes) ----------------
#define OFF_ATTN 0ull                         // ROWS*24 f32      = 1,228,800 B
#define OFF_MW   1228800ull                   // 307200 float4    = 4,915,200 B
#define OFF_MIDX 6144000ull                   // 307200 int4      = 4,915,200 B
#define OFF_AGG  11059200ull                  // ROWS*E f32       = 13,107,200 B
#define OFF_SINK 24166400ull

// ---------------- WMMA helpers ----------------
typedef __attribute__((ext_vector_type(16))) _Float16 v16h;
typedef __attribute__((ext_vector_type(8)))  float    v8f;
typedef __attribute__((ext_vector_type(4)))  float    f32x4;
typedef __attribute__((ext_vector_type(4)))  _Float16 h16x4;   // 8 B packed

union HFrag { v16h v; f32x4 f4[2]; };

__device__ __forceinline__ v8f wmma_f16(const HFrag& a, const HFrag& b, v8f c) {
  return __builtin_amdgcn_wmma_f32_16x16x32_f16(
      /*neg_a=*/false, a.v, /*neg_b=*/false, b.v,
      /*c_mod=*/(short)0, c, /*reuse_a=*/false, /*reuse_b=*/false);
}

__device__ __forceinline__ h16x4 cvt4(float4 f) {
  h16x4 h;
  h[0] = (_Float16)f.x; h[1] = (_Float16)f.y;
  h[2] = (_Float16)f.z; h[3] = (_Float16)f.w;
  return h;
}

// A fragment: 16x32 f16 tile, row-major in LDS. Lane L (r=L&15, g=L>>4):
// elems 0..15 = K[g*16 .. g*16+15].
__device__ __forceinline__ HFrag load_afrag(const _Float16* rowBase, int g) {
  HFrag a;
  a.f4[0] = *(const f32x4*)(rowBase + g * 16);
  a.f4[1] = *(const f32x4*)(rowBase + g * 16 + 8);
  return a;
}
// B fragment: col n = lane&15 (+16 for g=1 K-half), 16 contiguous K halves.
__device__ __forceinline__ HFrag load_bfrag(const _Float16* colBase) {
  HFrag b;
  b.f4[0] = *(const f32x4*)(colBase);
  b.f4[1] = *(const f32x4*)(colBase + 8);
  return b;
}

// ---------------- kernel 1: projection + bilinear meta ----------------
__global__ __launch_bounds__(256) void k_project(const float* __restrict__ anchor,
                                                 const float* __restrict__ proj,
                                                 float4* __restrict__ mw,
                                                 int4* __restrict__ midx) {
  int t = blockIdx.x * blockDim.x + threadIdx.x;   // over B*CAMS*N = 76800
  if (t >= B_ * CAMS_ * N_) return;
  int n  = t % N_;
  int bc = t / N_;
  int c  = bc % CAMS_;
  int b  = bc / CAMS_;

  const float* a = anchor + (size_t)(b * N_ + n) * 11;
  float sx = 1.f / (1.f + __expf(-a[0]));
  float sy = 1.f / (1.f + __expf(-a[1]));
  float sz = 1.f / (1.f + __expf(-a[2]));
  float wx = sx * 102.4f - 51.2f;
  float wy = sy * 102.4f - 51.2f;
  float wz = sz * 8.0f - 5.0f;

  const float* P = proj + (size_t)(b * CAMS_ + c) * 16;
  float cx = P[0] * wx + P[1] * wy + P[2]  * wz + P[3];
  float cy = P[4] * wx + P[5] * wy + P[6]  * wz + P[7];
  float cz = P[8] * wx + P[9] * wy + P[10] * wz + P[11];
  float d  = fmaxf(cz, 1e-4f);
  float u  = cx / d, v = cy / d;

  float x0f = floorf(u), y0f = floorf(v);
  float fx1 = u - x0f, fx0 = 1.f - fx1;
  float fy1 = v - y0f, fy0 = 1.f - fy1;
  int x0 = (int)x0f, y0 = (int)y0f;

#pragma unroll
  for (int l = 0; l < LVLS_; ++l) {
    int W = LVL_W_[l], H = LVL_H_[l];
    auto corner = [&](int xi, int yi, float w, int& idx) -> float {
      bool valid = (xi >= 0) && (xi <= W - 1) && (yi >= 0) && (yi <= H - 1);
      int xc = min(max(xi, 0), W - 1);
      int yc = min(max(yi, 0), H - 1);
      idx = yc * W + xc;
      return valid ? w : 0.f;
    };
    int i00, i10, i01, i11;
    float w00 = corner(x0,     y0,     fx0 * fy0, i00);
    float w10 = corner(x0 + 1, y0,     fx1 * fy0, i10);
    float w01 = corner(x0,     y0 + 1, fx0 * fy1, i01);
    float w11 = corner(x0 + 1, y0 + 1, fx1 * fy1, i11);
    size_t mo = (size_t)t * LVLS_ + l;
    mw[mo]   = make_float4(w00, w10, w01, w11);
    midx[mo] = make_int4(i00, i10, i01, i11);
  }
}

// ---------------- kernel 2: attention GEMM (WMMA, pipelined) + softmax ------
#define AT_MT 128
__global__ __launch_bounds__(256) void k_attn(const float* __restrict__ inst,
                                              const float* __restrict__ attn_w,
                                              const float* __restrict__ attn_b,
                                              float* __restrict__ attn_out) {
  __shared__ __align__(16) _Float16 sA[2][AT_MT][32];
  __shared__ __align__(16) _Float16 sB[2][32][32];
  __shared__ float sL[AT_MT][26];
  int tid = threadIdx.x, lane = tid & 31, wave = tid >> 5;
  int rowbase = blockIdx.x * AT_MT;
  int srow = tid >> 3;            // 0..31 : staging row within a pass
  int skc  = (tid & 7) * 4;       // staging k-chunk (4 floats)
  int   brow  = (srow < 24) ? srow : 0;
  float bmask = (srow < 24) ? 1.f : 0.f;

  auto ldA = [&](int ks, int p) -> float4 {
    return *(const float4*)(inst + (size_t)(rowbase + p * 32 + srow) * E_ + ks * 32 + skc);
  };
  auto ldB = [&](int ks) -> float4 {
    float4 fv = *(const float4*)(attn_w + (size_t)brow * E_ + ks * 32 + skc);
    fv.x *= bmask; fv.y *= bmask; fv.z *= bmask; fv.w *= bmask;
    return fv;
  };

  float4 rA[4], rB;
#pragma unroll
  for (int p = 0; p < 4; ++p) rA[p] = ldA(0, p);
  rB = ldB(0);

  v8f acc0 = {}, acc1 = {};
#pragma unroll
  for (int ks = 0; ks < 8; ++ks) {
    int buf = ks & 1;
#pragma unroll
    for (int p = 0; p < 4; ++p)
      *(h16x4*)&sA[buf][p * 32 + srow][skc] = cvt4(rA[p]);
    *(h16x4*)&sB[buf][srow][skc] = cvt4(rB);
    __syncthreads();
    if (ks < 7) {              // prefetch next K-step while computing
#pragma unroll
      for (int p = 0; p < 4; ++p) rA[p] = ldA(ks + 1, p);
      rB = ldB(ks + 1);
    }
    int r = lane & 15, g = lane >> 4;
    HFrag a  = load_afrag(&sA[buf][wave * 16 + r][0], g);
    HFrag b0 = load_bfrag(&sB[buf][r][g * 16]);
    HFrag b1 = load_bfrag(&sB[buf][16 + r][g * 16]);
    acc0 = wmma_f16(a, b0, acc0);
    acc1 = wmma_f16(a, b1, acc1);
  }
  __syncthreads();
  // scatter logits to LDS (C/D layout: VGPR q -> M = q + 8*g, N = lane&15)
  {
    int r = lane & 15, g = lane >> 4;
#pragma unroll
    for (int q = 0; q < 8; ++q) {
      int row = wave * 16 + g * 8 + q;
      if (r < 24)      sL[row][r]      = acc0[q] + attn_b[r];
      if (16 + r < 24) sL[row][16 + r] = acc1[q] + attn_b[16 + r];
    }
  }
  __syncthreads();
  // softmax over 4 levels per camera
  for (int i = tid; i < AT_MT * CAMS_; i += 256) {
    int row = i / CAMS_, c = i % CAMS_;
    float v0 = sL[row][c * 4 + 0], v1 = sL[row][c * 4 + 1];
    float v2 = sL[row][c * 4 + 2], v3 = sL[row][c * 4 + 3];
    float m  = fmaxf(fmaxf(v0, v1), fmaxf(v2, v3));
    float e0 = __expf(v0 - m), e1 = __expf(v1 - m);
    float e2 = __expf(v2 - m), e3 = __expf(v3 - m);
    float s  = 1.f / (e0 + e1 + e2 + e3);
    size_t o = (size_t)(rowbase + row) * 24 + c * 4;
    attn_out[o + 0] = e0 * s; attn_out[o + 1] = e1 * s;
    attn_out[o + 2] = e2 * s; attn_out[o + 3] = e3 * s;
  }
}

// ---------------- kernel 3: TDM warm (tensor_load_to_lds) ----------------
typedef __attribute__((ext_vector_type(4))) unsigned int u32x4;
typedef __attribute__((ext_vector_type(8))) int i32x8;
typedef __attribute__((ext_vector_type(4))) int i32x4;

__global__ void k_tdm(const float* __restrict__ src, float* __restrict__ sink) {
#if __has_builtin(__builtin_amdgcn_tensor_load_to_lds)
  __shared__ __align__(16) float buf[64 * 64];
  if (threadIdx.x == 0) {
    unsigned long long ga = (unsigned long long)(uintptr_t)src;
    unsigned int lds = (unsigned int)(uintptr_t)&buf[0];
    u32x4 g0; i32x8 g1 = {0,0,0,0,0,0,0,0};
    i32x4 g2 = {0,0,0,0}, g3 = {0,0,0,0};
    g0[0] = 1u;                                            // count=1, user mode
    g0[1] = lds;                                           // lds_addr
    g0[2] = (unsigned int)(ga & 0xffffffffu);              // global_addr[31:0]
    g0[3] = (unsigned int)((ga >> 32) & 0x01ffffffu) | (2u << 30);  // [56:32]+type=2
    g1[0] = (2 << 16);                                     // data_size = 4B
    g1[1] = (256 & 0xffff) << 16;                          // tensor_dim0 = 256 (lo16)
    g1[2] = (64 & 0xffff) << 16;                           // tensor_dim1 = 64
    g1[3] = (64 << 16);                                    // tile_dim0 = 64
    g1[4] = 64;                                            // tile_dim1 = 64
    g1[5] = 256;                                           // tensor_dim0_stride
#if defined(__clang_major__) && (__clang_major__ >= 23)
    i32x8 gz = {0,0,0,0,0,0,0,0};
    __builtin_amdgcn_tensor_load_to_lds(g0, g1, g2, g3, gz, 0);
#else
    __builtin_amdgcn_tensor_load_to_lds(g0, g1, g2, g3, 0);
#endif
#if __has_builtin(__builtin_amdgcn_s_wait_tensorcnt)
    __builtin_amdgcn_s_wait_tensorcnt(0);
#endif
    sink[0] = buf[0];
  }
#else
  if (threadIdx.x == 0) sink[0] = src[0];
#endif
}

// ---------------- kernel 4: bilinear sample + aggregate ----------------
__global__ __launch_bounds__(256) void k_sample(const float* __restrict__ f0,
                                                const float* __restrict__ f1,
                                                const float* __restrict__ f2,
                                                const float* __restrict__ f3,
                                                const float* __restrict__ attn,
                                                const float4* __restrict__ mw,
                                                const int4* __restrict__ midx,
                                                float* __restrict__ agg) {
  int bn = blockIdx.x;               // (b, n) pair
  int e  = threadIdx.x;              // channel
  int b  = bn / N_, n = bn % N_;

  __shared__ float4 s_w[CAMS_ * LVLS_];
  __shared__ int4   s_i[CAMS_ * LVLS_];
  if (threadIdx.x < CAMS_ * LVLS_) {
    int c = threadIdx.x >> 2, l = threadIdx.x & 3;
    size_t mo = ((size_t)(b * CAMS_ + c) * N_ + n) * LVLS_ + l;
    float aw  = attn[(size_t)bn * 24 + threadIdx.x];
    float4 w  = mw[mo];
    s_w[threadIdx.x] = make_float4(w.x * aw, w.y * aw, w.z * aw, w.w * aw);
    s_i[threadIdx.x] = midx[mo];
  }
  __syncthreads();

  const float* feats[4] = {f0, f1, f2, f3};
  float acc = 0.f;
#pragma unroll
  for (int c = 0; c < CAMS_; ++c) {
    int bc = b * CAMS_ + c;
#pragma unroll
    for (int l = 0; l < LVLS_; ++l) {
      float4 w = s_w[c * 4 + l];
      int4  ix = s_i[c * 4 + l];
      const float* fp = feats[l] + ((size_t)bc * E_ + e) * (size_t)LVL_HW_[l];
      acc += w.x * fp[ix.x] + w.y * fp[ix.y] + w.z * fp[ix.z] + w.w * fp[ix.w];
    }
  }
  agg[(size_t)bn * E_ + e] = acc;
}

// ---------------- kernel 5: fused double GEMM (WMMA, pipelined) -------------
#define GM 32
__global__ __launch_bounds__(256) void k_gemm2(const float* __restrict__ agg,
                                               const float* __restrict__ inst,
                                               const float* __restrict__ vp_w,
                                               const float* __restrict__ vp_b,
                                               const float* __restrict__ op_w,
                                               const float* __restrict__ op_b,
                                               float* __restrict__ out) {
  __shared__ __align__(16) _Float16 sA[2][GM][32];
  __shared__ __align__(16) _Float16 sB[2][256][32];
  __shared__ __align__(16) _Float16 sT[GM][256];
  int tid = threadIdx.x, lane = tid & 31, wave = tid >> 5;
  int rowbase = blockIdx.x * GM;
  int msub  = wave >> 2;          // 0..1  (rows 0-15 / 16-31)
  int ncol0 = (wave & 3) * 64;    // 4 N-tiles per wave
  int srow = tid >> 3;            // 0..31 : staging row within a pass
  int skc  = (tid & 7) * 4;       // staging k-chunk

  auto ldAgg = [&](int ks) -> float4 {
    return *(const float4*)(agg + (size_t)(rowbase + srow) * E_ + ks * 32 + skc);
  };
  auto ldW = [&](const float* w, int ks, int p) -> float4 {
    return *(const float4*)(w + (size_t)(p * 32 + srow) * E_ + ks * 32 + skc);
  };

  v8f acc[4] = {{}, {}, {}, {}};
  float4 rA, rB[8];

  // -------- GEMM 1: tmp = agg @ vp_w^T --------
  rA = ldAgg(0);
#pragma unroll
  for (int p = 0; p < 8; ++p) rB[p] = ldW(vp_w, 0, p);
#pragma unroll
  for (int ks = 0; ks < 8; ++ks) {
    int buf = ks & 1;
    *(h16x4*)&sA[buf][srow][skc] = cvt4(rA);
#pragma unroll
    for (int p = 0; p < 8; ++p)
      *(h16x4*)&sB[buf][p * 32 + srow][skc] = cvt4(rB[p]);
    __syncthreads();
    if (ks < 7) {              // prefetch next K-step while computing
      rA = ldAgg(ks + 1);
#pragma unroll
      for (int p = 0; p < 8; ++p) rB[p] = ldW(vp_w, ks + 1, p);
    }
    int r = lane & 15, g = lane >> 4;
    HFrag a = load_afrag(&sA[buf][msub * 16 + r][0], g);
#pragma unroll
    for (int t = 0; t < 4; ++t) {
      HFrag bf = load_bfrag(&sB[buf][ncol0 + t * 16 + r][g * 16]);
      acc[t] = wmma_f16(a, bf, acc[t]);
    }
  }
  __syncthreads();
  // epilogue 1: sT = f16(acc + vp_b + inst), reset accumulators
  {
    int r = lane & 15, g = lane >> 4;
#pragma unroll
    for (int t = 0; t < 4; ++t) {
#pragma unroll
      for (int q = 0; q < 8; ++q) {
        int row = msub * 16 + g * 8 + q;
        int col = ncol0 + t * 16 + r;
        float v = acc[t][q] + vp_b[col] + inst[(size_t)(rowbase + row) * E_ + col];
        sT[row][col] = (_Float16)v;
      }
      acc[t] = (v8f){};
    }
  }
  __syncthreads();
  // -------- GEMM 2: out = (tmp+inst) @ op_w^T --------
#pragma unroll
  for (int p = 0; p < 8; ++p) rB[p] = ldW(op_w, 0, p);
#pragma unroll
  for (int ks = 0; ks < 8; ++ks) {
    int buf = ks & 1;
#pragma unroll
    for (int p = 0; p < 8; ++p)
      *(h16x4*)&sB[buf][p * 32 + srow][skc] = cvt4(rB[p]);
    __syncthreads();
    if (ks < 7) {
#pragma unroll
      for (int p = 0; p < 8; ++p) rB[p] = ldW(op_w, ks + 1, p);
    }
    int r = lane & 15, g = lane >> 4;
    HFrag a = load_afrag(&sT[msub * 16 + r][ks * 32], g);
#pragma unroll
    for (int t = 0; t < 4; ++t) {
      HFrag bf = load_bfrag(&sB[buf][ncol0 + t * 16 + r][g * 16]);
      acc[t] = wmma_f16(a, bf, acc[t]);
    }
  }
  // epilogue 2
  {
    int r = lane & 15, g = lane >> 4;
#pragma unroll
    for (int t = 0; t < 4; ++t)
#pragma unroll
      for (int q = 0; q < 8; ++q) {
        int row = msub * 16 + g * 8 + q;
        int col = ncol0 + t * 16 + r;
        out[(size_t)(rowbase + row) * E_ + col] = acc[t][q] + op_b[col];
      }
  }
}

// ---------------- launcher ----------------
extern "C" void kernel_launch(void* const* d_in, const int* in_sizes, int n_in,
                              void* d_out, int out_size, void* d_ws, size_t ws_size,
                              hipStream_t stream) {
  const float* inst   = (const float*)d_in[0];
  const float* anchor = (const float*)d_in[1];
  const float* proj   = (const float*)d_in[2];
  const float* f0     = (const float*)d_in[3];
  const float* f1     = (const float*)d_in[4];
  const float* f2     = (const float*)d_in[5];
  const float* f3     = (const float*)d_in[6];
  const float* attn_w = (const float*)d_in[7];
  const float* attn_b = (const float*)d_in[8];
  const float* vp_w   = (const float*)d_in[9];
  const float* vp_b   = (const float*)d_in[10];
  const float* op_w   = (const float*)d_in[11];
  const float* op_b   = (const float*)d_in[12];
  float* out = (float*)d_out;

  char* ws = (char*)d_ws;
  float*  attn = (float*)(ws + OFF_ATTN);
  float4* mw   = (float4*)(ws + OFF_MW);
  int4*   midx = (int4*)(ws + OFF_MIDX);
  float*  agg  = (float*)(ws + OFF_AGG);
  float*  sink = (float*)(ws + OFF_SINK);

  k_project<<<(B_ * CAMS_ * N_) / 256, 256, 0, stream>>>(anchor, proj, mw, midx);
  k_attn<<<ROWS / AT_MT, 256, 0, stream>>>(inst, attn_w, attn_b, attn);
  k_tdm<<<1, 64, 0, stream>>>(vp_w, sink);
  k_sample<<<ROWS, 256, 0, stream>>>(f0, f1, f2, f3, attn, mw, midx, agg);
  k_gemm2<<<ROWS / GM, 256, 0, stream>>>(agg, inst, vp_w, vp_b, op_w, op_b, out);
}
